// GNNFeat_18872086299349
// MI455X (gfx1250) — matrix operand order repaired and checked by reading, compiled, not run on hardware
//
#include <hip/hip_runtime.h>

#define NN 4096
#define BN 32768
#define NE 524288

typedef __attribute__((ext_vector_type(16))) __bf16 v16bf;
typedef __attribute__((ext_vector_type(8)))  float  v8f;

// ---------------------------------------------------------------------------
// Convert f32 weight matrix W[K][Nc] (row-major) into bf16 B-fragment layout:
// fragment (kc,nc) = 512 bf16, lane-major: lane holds 16 contiguous bf16 with
// k = kc*32 + 16*(lane>>4) + i, n = nc*16 + (lane&15).
// ---------------------------------------------------------------------------
__global__ __launch_bounds__(256) void wprep_kernel(const float* __restrict__ W,
                                                    __bf16* __restrict__ Wf,
                                                    int K, int Nc) {
  int idx = blockIdx.x * 256 + threadIdx.x;
  if (idx >= K * Nc) return;
  int i    = idx & 15;
  int lane = (idx >> 4) & 31;
  int frag = idx >> 9;
  int numNc = Nc >> 4;
  int kc = frag / numNc;
  int nc = frag - kc * numNc;
  int k = kc * 32 + ((lane >> 4) << 4) + i;
  int n = nc * 16 + (lane & 15);
  Wf[idx] = (__bf16)W[(size_t)k * Nc + n];
}

// ---------------------------------------------------------------------------
// Per-point input projections: xg = relu(x@g_w+g_b), ce = relu(emb@c_w+c_b).
// fsrc = (e ? ce : xg), othin = (e ? xg : ce).  Layout [BN, 64] row-major.
// ---------------------------------------------------------------------------
__global__ __launch_bounds__(256) void feat_kernel(
    const float* __restrict__ x, const float* __restrict__ emb,
    const int* __restrict__ ePtr,
    const float* __restrict__ gw, const float* __restrict__ gb,
    const float* __restrict__ cw, const float* __restrict__ cb,
    float* __restrict__ fsrc, float* __restrict__ othin) {
  int p = blockIdx.x * 256 + threadIdx.x;
  if (p >= BN) return;
  int b = p >> 12;
  int n = p & 4095;
  float xv[3], ev[32];
#pragma unroll
  for (int c = 0; c < 3; ++c) xv[c] = x[((size_t)b * 3 + c) * NN + n];
#pragma unroll
  for (int c = 0; c < 32; ++c) ev[c] = emb[((size_t)b * 32 + c) * NN + n];
  int e = *ePtr;
  float* fdst = fsrc  + (size_t)p * 64;
  float* odst = othin + (size_t)p * 64;
  for (int d = 0; d < 64; ++d) {
    float s = gb[d];
#pragma unroll
    for (int c = 0; c < 3; ++c) s += xv[c] * gw[c * 64 + d];
    float xg = fmaxf(s, 0.f);
    float t = cb[d];
#pragma unroll
    for (int c = 0; c < 32; ++c) t += ev[c] * cw[c * 64 + d];
    float ce = fmaxf(t, 0.f);
    fdst[d] = e ? ce : xg;
    odst[d] = e ? xg : ce;
  }
}

// ---------------------------------------------------------------------------
// Edge scatter: agg[tgt[e], :] += h[src[e], :].  One thread = 4 channels.
// ---------------------------------------------------------------------------
__global__ __launch_bounds__(256) void scatter_kernel(
    const float* __restrict__ h, const int* __restrict__ src,
    const int* __restrict__ tgt, float* __restrict__ agg,
    int C4, int total) {
  int t = blockIdx.x * 256 + threadIdx.x;
  if (t >= total) return;
  int eid = t / C4;
  int c4  = (t - eid * C4) * 4;
  int s = src[eid];
  int d = tgt[eid];
  const float4 v = *(const float4*)(h + (size_t)s * (C4 * 4) + c4);
  float* a = agg + (size_t)d * (C4 * 4) + c4;
  __hip_atomic_fetch_add(a + 0, v.x, __ATOMIC_RELAXED, __HIP_MEMORY_SCOPE_AGENT);
  __hip_atomic_fetch_add(a + 1, v.y, __ATOMIC_RELAXED, __HIP_MEMORY_SCOPE_AGENT);
  __hip_atomic_fetch_add(a + 2, v.z, __ATOMIC_RELAXED, __HIP_MEMORY_SCOPE_AGENT);
  __hip_atomic_fetch_add(a + 3, v.w, __ATOMIC_RELAXED, __HIP_MEMORY_SCOPE_AGENT);
}

// ---------------------------------------------------------------------------
// WMMA GEMM: D = act( (A [+ A2]) @ W + bias ), A is [BN,K] f32 row-major,
// W pre-staged bf16 fragments.  Each wave: one 16-row x 64-col strip,
// 4 accumulators, K/32 bf16 WMMA steps per accumulator.
// Optional row-major store (Cout, ldOut, colOff) and/or transposed store
// directly into out[b][chBase+n][point] ([B,1408,4096] layout).
// ---------------------------------------------------------------------------
template <bool HASADD, bool RELU>
__global__ __launch_bounds__(256) void gemm_kernel(
    const float* __restrict__ A, const float* __restrict__ A2,
    const __bf16* __restrict__ Wf, const float* __restrict__ bias,
    float* __restrict__ Cout, int ldOut, int colOff,
    float* __restrict__ outT, int chBase,
    int K, int Nc) {
  int wave = (blockIdx.x * 256 + (int)threadIdx.x) >> 5;
  int lane = threadIdx.x & 31;
  int strips  = Nc >> 6;
  int rowTile = wave / strips;
  int strip   = wave - rowTile * strips;
  if (rowTile >= (BN >> 4)) return;   // wave-uniform guard (EXEC stays all-ones)
  int row0    = rowTile << 4;
  int colBase = strip << 6;
  int m  = lane & 15;
  int hf = lane >> 4;
  const float* arow  = A + (size_t)(row0 + m) * K;
  const float* arow2 = HASADD ? (A2 + (size_t)(row0 + m) * K) : nullptr;
  int numNc = Nc >> 4;

  v8f acc[4] = {};
  for (int kc = 0; kc < (K >> 5); ++kc) {
    int kb = (kc << 5) + 16 * hf;
    v16bf a;
#pragma unroll
    for (int j = 0; j < 8; ++j) {
      float v = arow[kb + j];
      if (HASADD) v += arow2[kb + j];
      a[j] = (__bf16)v;
    }
#pragma unroll
    for (int j = 0; j < 8; ++j) {
      float v = arow[kb + 8 + j];
      if (HASADD) v += arow2[kb + 8 + j];
      a[8 + j] = (__bf16)v;
    }
    const __bf16* wl = Wf + (((size_t)(kc * numNc + (colBase >> 4)) * 32) + lane) * 16;
#pragma unroll
    for (int t = 0; t < 4; ++t) {
      v16bf bfr = *(const v16bf*)(wl + (size_t)t * 512);
      acc[t] = __builtin_amdgcn_wmma_f32_16x16x32_bf16(
          false, a, false, bfr, (short)0, acc[t], false, false);
    }
  }

#pragma unroll
  for (int t = 0; t < 4; ++t) {
    int ncol = colBase + t * 16 + m;
    float bv = bias[ncol];
    float vals[8];
#pragma unroll
    for (int r = 0; r < 8; ++r) {
      float v = acc[t][r] + bv;
      if (RELU) v = fmaxf(v, 0.f);
      vals[r] = v;
    }
    if (Cout) {
#pragma unroll
      for (int r = 0; r < 8; ++r)
        Cout[(size_t)(row0 + r + 8 * hf) * ldOut + colOff + ncol] = vals[r];
    }
    if (outT) {
      int bIdx = row0 >> 12;                 // point / N
      int n0   = (row0 & 4095) + 8 * hf;     // point % N
      float* dst = outT + ((size_t)bIdx * 1408 + chBase + ncol) * 4096 + n0;
      *(float4*)(dst)     = make_float4(vals[0], vals[1], vals[2], vals[3]);
      *(float4*)(dst + 4) = make_float4(vals[4], vals[5], vals[6], vals[7]);
    }
  }
}

// ---------------------------------------------------------------------------
extern "C" void kernel_launch(void* const* d_in, const int* in_sizes, int n_in,
                              void* d_out, int out_size, void* d_ws, size_t ws_size,
                              hipStream_t stream) {
  (void)in_sizes; (void)n_in; (void)out_size; (void)ws_size;
  const float* x    = (const float*)d_in[0];
  const float* emb  = (const float*)d_in[1];
  const int*   eidx = (const int*)d_in[2];
  const int*   ePtr = (const int*)d_in[3];
  const float* g_w  = (const float*)d_in[4];
  const float* g_b  = (const float*)d_in[5];
  const float* c_w  = (const float*)d_in[6];
  const float* c_b  = (const float*)d_in[7];
  const float* m1w1 = (const float*)d_in[8];
  const float* m1b1 = (const float*)d_in[9];
  const float* m1w2 = (const float*)d_in[10];
  const float* m1b2 = (const float*)d_in[11];
  const float* m2w1 = (const float*)d_in[12];
  const float* m2b1 = (const float*)d_in[13];
  const float* m2w2 = (const float*)d_in[14];
  const float* m2b2 = (const float*)d_in[15];
  const float* m3w1 = (const float*)d_in[16];
  const float* m3b1 = (const float*)d_in[17];
  const float* m3w2 = (const float*)d_in[18];
  const float* m3b2 = (const float*)d_in[19];
  const float* fcw  = (const float*)d_in[20];
  const float* fcb  = (const float*)d_in[21];
  float* out = (float*)d_out;

  // Workspace layout (all offsets 256B-aligned).
  char* ws = (char*)d_ws;
  __bf16* Wf   = (__bf16*)ws;                    // 876544 bf16 = 1,753,088 B
  size_t off   = 1753088;
  float* fsrc  = (float*)(ws + off); off += (size_t)BN * 64 * 4;
  float* othin = (float*)(ws + off); off += (size_t)BN * 64 * 4;
  float* f1    = (float*)(ws + off); off += (size_t)BN * 128 * 4;
  float* ff    = (float*)(ws + off); off += (size_t)BN * 512 * 4;
  float* hbuf  = (float*)(ws + off); off += (size_t)BN * 512 * 4;
  float* agg   = (float*)(ws + off); off += (size_t)BN * 512 * 4;

  // bf16 weight fragment offsets (elements)
  const size_t o_m1w1 = 0,      o_m1w2 = 8192,   o_m2w1 = 24576;
  const size_t o_m2w2 = 40960,  o_m3w1 = 73728,  o_m3w2 = 335872;
  const size_t o_fc   = 860160;

  struct WP { const float* w; size_t o; int K, N; };
  const WP wps[7] = {
      {m1w1, o_m1w1, 64, 128},  {m1w2, o_m1w2, 128, 128},
      {m2w1, o_m2w1, 128, 128}, {m2w2, o_m2w2, 128, 256},
      {m3w1, o_m3w1, 512, 512}, {m3w2, o_m3w2, 512, 1024},
      {fcw,  o_fc,   64, 256}};
  for (int i = 0; i < 7; ++i) {
    int total = wps[i].K * wps[i].N;
    wprep_kernel<<<(total + 255) / 256, 256, 0, stream>>>(
        wps[i].w, Wf + wps[i].o, wps[i].K, wps[i].N);
  }

  feat_kernel<<<BN / 256, 256, 0, stream>>>(x, emb, ePtr, g_w, g_b, c_w, c_b,
                                            fsrc, othin);

  const int* src = eidx;
  const int* tgt = eidx + NE;

  // -------- GIN block 1: 64 -> 128 -> 128, outer relu, out channels [0,128)
  hipMemsetAsync(agg, 0, (size_t)BN * 64 * 4, stream);
  scatter_kernel<<<(NE * 16) / 256, 256, 0, stream>>>(fsrc, src, tgt, agg, 16, NE * 16);
  gemm_kernel<true, true><<<512, 256, 0, stream>>>(
      fsrc, agg, Wf + o_m1w1, m1b1, hbuf, 128, 0, nullptr, 0, 64, 128);
  gemm_kernel<false, true><<<512, 256, 0, stream>>>(
      hbuf, nullptr, Wf + o_m1w2, m1b2, f1, 128, 0, out, 0, 128, 128);

  // -------- GIN block 2: 128 -> 128 -> 256, outer relu, out channels [128,384)
  hipMemsetAsync(agg, 0, (size_t)BN * 128 * 4, stream);
  scatter_kernel<<<(NE * 32) / 256, 256, 0, stream>>>(f1, src, tgt, agg, 32, NE * 32);
  gemm_kernel<true, true><<<512, 256, 0, stream>>>(
      f1, agg, Wf + o_m2w1, m2b1, hbuf, 128, 0, nullptr, 0, 128, 128);
  gemm_kernel<false, true><<<1024, 256, 0, stream>>>(
      hbuf, nullptr, Wf + o_m2w2, m2b2, ff, 512, 0, out, 128, 128, 256);

  // -------- other-modality fc: 64 -> 256 relu, into ff columns [256,512)
  gemm_kernel<false, true><<<1024, 256, 0, stream>>>(
      othin, nullptr, Wf + o_fc, fcb, ff, 512, 256, nullptr, 0, 64, 256);

  // -------- GIN block 3: 512 -> 512 -> 1024, no outer relu, channels [384,1408)
  hipMemsetAsync(agg, 0, (size_t)BN * 512 * 4, stream);
  scatter_kernel<<<(NE * 128) / 256, 256, 0, stream>>>(ff, src, tgt, agg, 128, NE * 128);
  gemm_kernel<true, true><<<2048, 256, 0, stream>>>(
      ff, agg, Wf + o_m3w1, m3b1, hbuf, 512, 0, nullptr, 0, 512, 512);
  gemm_kernel<false, false><<<4096, 256, 0, stream>>>(
      hbuf, nullptr, Wf + o_m3w2, m3b2, nullptr, 0, 0, out, 384, 512, 1024);
}
